// RefinePrior_188978561657
// MI455X (gfx1250) — matrix operand order repaired and checked by reading, compiled, not run on hardware
//
#include <hip/hip_runtime.h>
#include <stdint.h>

typedef __attribute__((ext_vector_type(16))) __bf16 v16bf;
typedef __attribute__((ext_vector_type(8)))  float  v8f;
typedef __attribute__((ext_vector_type(4)))  unsigned int u32x4;
typedef __attribute__((ext_vector_type(4)))  float  f32x4;
typedef unsigned int   u32;
typedef unsigned short u16;

#define H_ 16
#define D_ 1024
#define HD_ 64
#define N_ 16
#define B_ 4096
#define M_ (B_ * N_)   // 65536 rows

union ABu { u32x4 u4[2]; v16bf v; };

__device__ __forceinline__ u16 f2bf(float f) {
  u32 u = __float_as_uint(f);
  u32 r = u + 0x7FFFu + ((u >> 16) & 1u);   // round-to-nearest-even
  return (u16)(r >> 16);
}

__device__ __forceinline__ v8f wmma_bf16(const ABu& a, const ABu& b, v8f c) {
  return __builtin_amdgcn_wmma_f32_16x16x32_bf16(false, a.v, false, b.v,
                                                 (short)0, c, false, false);
}

// Issue one 16-byte async DMA chunk: global -> LDS (tracked by ASYNCcnt).
__device__ __forceinline__ void async_b128(u32* lds_dst, const u32* gsrc) {
  u32 lds = (u32)(uintptr_t)lds_dst;                     // addr[31:0] = LDS byte addr
  unsigned long long ga = (unsigned long long)(uintptr_t)gsrc;
  asm volatile("global_load_async_to_lds_b128 %0, %1, off"
               :: "v"(lds), "v"(ga) : "memory");
}
__device__ __forceinline__ void wait_async0() {
  asm volatile("s_wait_asynccnt 0x0" ::: "memory");
}

// ---------------------------------------------------------------- cvt x->bf16
__global__ void cvt_bf16_kernel(const float* __restrict__ x, u16* __restrict__ y) {
  size_t i = ((size_t)blockIdx.x * blockDim.x + threadIdx.x) * 4;
  f32x4 v = *(const f32x4*)(x + i);
  u32* dst = (u32*)(y + i);
  dst[0] = (u32)f2bf(v.x) | ((u32)f2bf(v.y) << 16);
  dst[1] = (u32)f2bf(v.z) | ((u32)f2bf(v.w) << 16);
}

// ------------------------------------------------- pack W into WMMA B layout
// dst dword t: v=t&7, L=(t>>3)&31, tile=t>>8, kt=tile&31, nt=tile>>5
// value = pack(W[kt*32 + (v>>2)*16 + g*8 + 2*(v&3)][nt*16 + (L&15)], W[k+1][n])
__global__ void pack_w_kernel(const float* __restrict__ W, u32* __restrict__ dst) {
  int t = blockIdx.x * 256 + threadIdx.x;
  int v = t & 7, L = (t >> 3) & 31, tile = t >> 8;
  int kt = tile & 31, nt = tile >> 5;
  int g = L >> 4, n = nt * 16 + (L & 15);
  int k = kt * 32 + (v >> 2) * 16 + g * 8 + 2 * (v & 3);
  u32 lo = f2bf(W[(size_t)k * D_ + n]);
  u32 hi = f2bf(W[(size_t)(k + 1) * D_ + n]);
  dst[t] = lo | (hi << 16);
}

// -------------------------------------------- cooperative GEMM inner loop
// A: row-major bf16, 16 rows per wave (8 waves = 8 M-tiles), stride D_.
// B: NT packed tiles shared by the whole block; staged into LDS with async
//    DMA, double-buffered (pointer swap), consumed via ds_read_b128.
template<int NT>
__device__ __forceinline__ void gemm_block(const u16* __restrict__ Arows,
                                           const u32* __restrict__ Bp,
                                           int nt0, u32* sB /* [2][NT*256] */,
                                           v8f acc[NT]) {
  constexpr int NSTG = NT / 4;              // 16B async chunks per thread/stage
  const int tid  = threadIdx.x;
  const int lane = tid & 31;
  const int m = lane & 15, g = lane >> 4;
  const u16* abase = Arows + (size_t)m * D_ + g * 8;
  const int jj = tid >> 6, cc = tid & 63;
  const u32* bsrc[NSTG];
#pragma unroll
  for (int p = 0; p < NSTG; ++p)            // source of this thread's chunks
    bsrc[p] = Bp + (size_t)(nt0 + jj + 4 * p) * 32 * 256 + cc * 4;

  u32* cur = sB;
  u32* nxt = sB + NT * 256;

  // stage kt = 0 into cur
#pragma unroll
  for (int p = 0; p < NSTG; ++p)
    async_b128(cur + tid * 4 + p * 1024, bsrc[p]);
  wait_async0();
  __syncthreads();

#pragma unroll 1                            // keep acc tuples pinned (no rotation)
  for (int kt = 0; kt < 32; ++kt) {
    if (kt < 31) {                          // kick DMA for next k-step
#pragma unroll
      for (int p = 0; p < NSTG; ++p)
        async_b128(nxt + tid * 4 + p * 1024, bsrc[p] + (size_t)(kt + 1) * 256);
    }
    __builtin_prefetch(abase + (kt + 2) * 32, 0, 3);   // near prefetch, A stream

    ABu a;
    const u16* ap = abase + kt * 32;
    a.u4[0] = *(const u32x4*)(ap);
    a.u4[1] = *(const u32x4*)(ap + 16);
#pragma unroll
    for (int j = 0; j < NT; ++j) {
      ABu b;
      const u32* bp = cur + j * 256 + lane * 8;
      b.u4[0] = *(const u32x4*)(bp);
      b.u4[1] = *(const u32x4*)(bp + 4);
      acc[j] = wmma_bf16(a, b, acc[j]);
    }
    wait_async0();      // own DMA for nxt done (ran concurrently with WMMAs)
    __syncthreads();    // everyone done reading cur + all DMAs visible
    u32* t = cur; cur = nxt; nxt = t;
  }
}

// ------------------------------------------------------------- QKV GEMM
// grid (24, 512), block 256. wave w handles m-tile (=batch b) mg*8+w,
// all waves share n-tiles ng*8 .. ng*8+7 of the 3072-column [Wq|Wk|Wv].
__global__ void qkv_gemm_kernel(const u16* __restrict__ xbf,
                                const u32* __restrict__ Wp,
                                const float* __restrict__ bq,
                                const float* __restrict__ bk,
                                const float* __restrict__ bv,
                                u16* __restrict__ q_rm,
                                u16* __restrict__ k_rm,
                                u16* __restrict__ v_t) {
  __shared__ u32 sB[2 * 8 * 256];           // 16 KB double buffer
  const int lane = threadIdx.x & 31, w = threadIdx.x >> 5;
  const int ng = blockIdx.x, mg = blockIdx.y;
  const int b = mg * 8 + w;                 // batch index == m-tile
  const int nt0 = ng * 8;
  v8f acc[8] = {};
  gemm_block<8>(xbf + (size_t)b * 16 * D_, Wp, nt0, sB, acc);

  const int n = lane & 15, g = lane >> 4;
#pragma unroll
  for (int j = 0; j < 8; ++j) {
    const int col = (nt0 + j) * 16 + n;                    // 0..3071
    const float bias = (col < 1024) ? bq[col]
                     : (col < 2048) ? bk[col - 1024] : bv[col - 2048];
    const int proj = col >> 10, d = col & 1023;
    const int h = d >> 6, f = d & 63;
    const size_t base = ((size_t)b * H_ + h) * (N_ * HD_); // per-(b,h) 1024
#pragma unroll
    for (int r = 0; r < 8; ++r) {
      const int m = r + g * 8;
      const u16 us = f2bf(acc[j][r] + bias);
      if (proj == 0)      q_rm[base + m * HD_ + f] = us;   // [16][64] row-major
      else if (proj == 1) k_rm[base + m * HD_ + f] = us;   // [16][64] row-major
      else                v_t [base + f * N_  + m] = us;   // [64][16] transposed
    }
  }
}

// ------------------------------------------------------------- attention
// one wave per (b,h). grid 8192, block 256 (8 waves).
__global__ void attn_kernel(const u16* __restrict__ q_rm,
                            const u16* __restrict__ k_rm,
                            const u16* __restrict__ v_t,
                            const float* __restrict__ pos,
                            u16* __restrict__ ctx) {
  __shared__ u16 sp[8][16 * 32];                 // probs, K padded 16->32
  const int lane = threadIdx.x & 31, w = threadIdx.x >> 5;
  const int idx = blockIdx.x * 8 + w;            // b*H + h
  const int b = idx >> 4, h = idx & 15;
  const size_t base = (size_t)idx * (N_ * HD_);
  const int t = lane & 15, g = lane >> 4;

  // zero the whole probs tile (cols 16..31 must be 0 for the padded K)
  u32* spz = (u32*)&sp[w][0];                    // 256 dwords
#pragma unroll
  for (int i = 0; i < 8; ++i) spz[lane + 32 * i] = 0;

  // scores = q @ k^T  (K = 64 -> two WMMAs)
  v8f s = {};
#pragma unroll
  for (int kt = 0; kt < 2; ++kt) {
    ABu a, bm;
    const u16* qp = q_rm + base + t * HD_ + kt * 32 + g * 8;
    a.u4[0] = *(const u32x4*)(qp);
    a.u4[1] = *(const u32x4*)(qp + 16);
    const u16* kp = k_rm + base + t * HD_ + kt * 32 + g * 8; // row n=t of K
    bm.u4[0] = *(const u32x4*)(kp);
    bm.u4[1] = *(const u32x4*)(kp + 16);
    s = wmma_bf16(a, bm, s);
  }

  // softmax per row (row m = r+8g lives in one 16-lane half, same VGPR r)
  float p8[8];
#pragma unroll
  for (int r = 0; r < 8; ++r) {
    const int mr = r + g * 8;
    float sv = s[r] * 0.125f + pos[h * 256 + mr * 16 + t];
    float mx = sv;
    mx = fmaxf(mx, __shfl_xor(mx, 1, 32));
    mx = fmaxf(mx, __shfl_xor(mx, 2, 32));
    mx = fmaxf(mx, __shfl_xor(mx, 4, 32));
    mx = fmaxf(mx, __shfl_xor(mx, 8, 32));
    float e = __expf(sv - mx);
    float sum = e;
    sum += __shfl_xor(sum, 1, 32);
    sum += __shfl_xor(sum, 2, 32);
    sum += __shfl_xor(sum, 4, 32);
    sum += __shfl_xor(sum, 8, 32);
    p8[r] = e / sum;
  }
#pragma unroll
  for (int r = 0; r < 8; ++r) sp[w][(r + g * 8) * 32 + t] = f2bf(p8[r]);
  __syncthreads();

  // ctx = probs @ V : A = padded probs from LDS, B = v_t (lanes 16..31 -> 0)
  ABu a;
  const u16* app = &sp[w][t * 32 + g * 16];
  a.u4[0] = *(const u32x4*)(app);
  a.u4[1] = *(const u32x4*)(app + 8);
#pragma unroll
  for (int ft = 0; ft < 4; ++ft) {
    ABu bv_;
    if (g == 0) {
      const u16* vp = v_t + base + (size_t)(ft * 16 + t) * 16;
      bv_.u4[0] = *(const u32x4*)(vp);
      bv_.u4[1] = *(const u32x4*)(vp + 8);
    } else {
      bv_.u4[0] = (u32x4)(0u);
      bv_.u4[1] = (u32x4)(0u);
    }
    v8f c = {};
    c = wmma_bf16(a, bv_, c);
#pragma unroll
    for (int r = 0; r < 8; ++r) {
      const int mr = r + g * 8;
      ctx[(size_t)(b * 16 + mr) * D_ + h * 64 + ft * 16 + t] = f2bf(c[r]);
    }
  }
}

// ------------------------------------------------------------- output GEMM
__global__ void out_gemm_kernel(const u16* __restrict__ ctx,
                                const u32* __restrict__ Wop,
                                const float* __restrict__ bo,
                                float* __restrict__ out) {
  __shared__ u32 sB[2 * 8 * 256];           // 16 KB double buffer
  const int lane = threadIdx.x & 31, w = threadIdx.x >> 5;
  const int ng = blockIdx.x, mg = blockIdx.y;
  const int mt = mg * 8 + w;
  const int nt0 = ng * 8;
  v8f acc[8] = {};
  gemm_block<8>(ctx + (size_t)mt * 16 * D_, Wop, nt0, sB, acc);

  const int n = lane & 15, g = lane >> 4;
#pragma unroll
  for (int j = 0; j < 8; ++j) {
    const int col = (nt0 + j) * 16 + n;
    const float bias = bo[col];
#pragma unroll
    for (int r = 0; r < 8; ++r) {
      const int m = r + g * 8;
      out[(size_t)(mt * 16 + m) * D_ + col] = acc[j][r] + bias;
    }
  }
}

// ---------------------------------------------------------------- launcher
extern "C" void kernel_launch(void* const* d_in, const int* in_sizes, int n_in,
                              void* d_out, int out_size, void* d_ws, size_t ws_size,
                              hipStream_t stream) {
  const float* x   = (const float*)d_in[0];
  const float* Wq  = (const float*)d_in[1];
  const float* bq  = (const float*)d_in[2];
  const float* Wk  = (const float*)d_in[3];
  const float* bk  = (const float*)d_in[4];
  const float* Wv  = (const float*)d_in[5];
  const float* bv  = (const float*)d_in[6];
  const float* Wo  = (const float*)d_in[7];
  const float* bo  = (const float*)d_in[8];
  const float* pos = (const float*)d_in[9];

  char* ws = (char*)d_ws;
  const size_t MB = 1024 * 1024;
  u16* xbf  = (u16*)(ws + 0);             // 128 MB  (aliased by ctx later)
  u16* q_rm = (u16*)(ws + 128 * MB);      // 128 MB
  u16* k_rm = (u16*)(ws + 256 * MB);      // 128 MB
  u16* v_t  = (u16*)(ws + 384 * MB);      // 128 MB
  u32* Wp   = (u32*)(ws + 512 * MB);      //   6 MB (Wq|Wk|Wv packed)
  u32* Wop  = (u32*)(ws + 518 * MB);      //   2 MB
  u16* ctx  = xbf;                        // reuse: x last read in qkv_gemm

  // 1) x -> bf16 row-major
  cvt_bf16_kernel<<<M_ * D_ / (256 * 4), 256, 0, stream>>>(x, xbf);
  // 2) weights -> WMMA B-operand packed layout
  pack_w_kernel<<<2048, 256, 0, stream>>>(Wq, Wp);
  pack_w_kernel<<<2048, 256, 0, stream>>>(Wk, Wp + 524288);
  pack_w_kernel<<<2048, 256, 0, stream>>>(Wv, Wp + 1048576);
  pack_w_kernel<<<2048, 256, 0, stream>>>(Wo, Wop);
  // 3) fused QKV projection (async-LDS staged B, double-buffered, NT=8)
  qkv_gemm_kernel<<<dim3(24, 512), 256, 0, stream>>>(xbf, Wp, bq, bk, bv,
                                                     q_rm, k_rm, v_t);
  // 4) per-head attention (QK^T + softmax + PV), one wave per (b,h)
  attn_kernel<<<B_ * H_ / 8, 256, 0, stream>>>(q_rm, k_rm, v_t, pos, ctx);
  // 5) output projection
  out_gemm_kernel<<<dim3(8, 512), 256, 0, stream>>>(ctx, Wop, bo, (float*)d_out);
}
